// CausalVAELayer_43568148251097
// MI455X (gfx1250) — matrix operand order repaired and checked by reading, not compiled, not run on hardware
//
#include <hip/hip_runtime.h>
#include <math.h>

// ---------------------------------------------------------------------------
// Problem constants (match reference)
// ---------------------------------------------------------------------------
static constexpr int Bc  = 8;
static constexpr int Sc  = 2048;
static constexpr int Dc  = 1024;
static constexpr int Cc  = 16;
static constexpr int CDc = 64;
static constexpr int Lc  = Cc * CDc;          // 1024
static constexpr int TWO_L = 2 * Lc;          // 2048
static constexpr int Mc  = Bc * Sc;           // 16384 tokens
static constexpr float LN_EPS = 1e-5f;
static constexpr float ALPHA = 0.3f;

// ---------------------------------------------------------------------------
// Vector types for WMMA
// ---------------------------------------------------------------------------
typedef __attribute__((ext_vector_type(16))) __bf16 v16bf;
typedef __attribute__((ext_vector_type(8)))  __bf16 v8bf;
typedef __attribute__((ext_vector_type(8)))  float  v8f;

__device__ __forceinline__ float softplusf(float x) {
    return (x > 20.f) ? x : log1pf(expf(x));
}

// Shared-memory block reduction (blockDim.x == 256), deterministic tree.
__device__ __forceinline__ float blockReduce256(float v, float* red) {
    int t = threadIdx.x;
    red[t] = v; __syncthreads();
    for (int s = 128; s > 0; s >>= 1) {
        if (t < s) red[t] += red[t + s];
        __syncthreads();
    }
    float r = red[0]; __syncthreads();
    return r;
}

// ---------------------------------------------------------------------------
// f32 -> bf16 cast (grid-strided)
// ---------------------------------------------------------------------------
__global__ void cvt_f32_bf16(const float* __restrict__ in, __bf16* __restrict__ out, long n) {
    long i = (long)blockIdx.x * blockDim.x + threadIdx.x;
    long stride = (long)gridDim.x * blockDim.x;
    for (; i < n; i += stride) out[i] = (__bf16)in[i];
}

// ---------------------------------------------------------------------------
// Generic (optionally batched) bf16 GEMM:  out[m,n] = sum_k A[m,k] * W[n,k] + bias[n]
//   A: [M,K] row-major bf16 (lda == K), W: [N,K] row-major bf16 (torch Linear layout)
//   Wave tile: 32 (M) x 64 (N) -> 2 A-frags, 4 B-frags, 8 WMMAs per K-step of 32.
//   Per-lane stream pointers are advanced by 32 elements/step so the inner loop
//   has zero address arithmetic (global_load_b128 base+imm only).
//   Batch index = blockIdx.y (strides in elements).
// ---------------------------------------------------------------------------
__global__ void __launch_bounds__(256, 1)
gemm_bf16_wmma(const __bf16* __restrict__ A,
               const __bf16* __restrict__ W,
               const float*  __restrict__ bias,
               float*  __restrict__ outF,
               __bf16* __restrict__ outB,
               int M, int N, int K,
               int ldo, int ldob, int obOff,
               long strideA, long strideW, long strideBias,
               long strideOutF, int obBatchOff) {
    const int b = blockIdx.y;
    A += (long)b * strideA;
    W += (long)b * strideW;
    if (bias) bias += (long)b * strideBias;
    if (outF) outF += (long)b * strideOutF;
    const int obTot = obOff + b * obBatchOff;

    const int lane  = threadIdx.x & 31;
    const int wave  = (blockIdx.x * blockDim.x + threadIdx.x) >> 5;
    const int TM = M >> 5;           // 32-row wave tiles
    const int TN = N >> 6;           // 64-col wave tiles
    if (wave >= TM * TN) return;

    const int mt = wave % TM;        // consecutive waves in a block share the B tile
    const int nt = wave / TM;
    const int m0 = mt << 5;
    const int n0 = nt << 6;

    const int half = lane >> 4;      // 0: lanes 0-15, 1: lanes 16-31
    const int lm   = lane & 15;

    // A fragment streams: lane lm holds row m0+lm (and m0+16+lm);
    // lanes<16 take K[k0..k0+15], lanes>=16 take K[k0+16..k0+31] (matches B).
    const __bf16* pA0 = A + (long)(m0 + lm) * K + half * 16;
    const __bf16* pA1 = pA0 + (long)16 * K;
    // B fragment streams: lane lm holds column n; lanes<16 take K[k0..k0+15],
    // lanes>=16 take K[k0+16..k0+31] (one contiguous 32B region per lane).
    const __bf16* pB0 = W + (long)(n0 + lm) * K + half * 16;
    const __bf16* pB1 = pB0 + (long)16 * K;
    const __bf16* pB2 = pB0 + (long)32 * K;
    const __bf16* pB3 = pB0 + (long)48 * K;

    v8f acc[2][4];
    #pragma unroll
    for (int s = 0; s < 2; ++s)
        #pragma unroll
        for (int j = 0; j < 4; ++j)
            acc[s][j] = (v8f){0.f,0.f,0.f,0.f,0.f,0.f,0.f,0.f};

    for (int k0 = 0; k0 < K; k0 += 32) {
        v16bf aF[2];
        {
            v8bf lo0 = *(const v8bf*)(pA0);
            v8bf hi0 = *(const v8bf*)(pA0 + 8);
            v8bf lo1 = *(const v8bf*)(pA1);
            v8bf hi1 = *(const v8bf*)(pA1 + 8);
            #pragma unroll
            for (int i = 0; i < 8; ++i) {
                aF[0][i] = lo0[i]; aF[0][i + 8] = hi0[i];
                aF[1][i] = lo1[i]; aF[1][i + 8] = hi1[i];
            }
        }
        const v16bf b0 = *(const v16bf*)(pB0);
        const v16bf b1 = *(const v16bf*)(pB1);
        const v16bf b2 = *(const v16bf*)(pB2);
        const v16bf b3 = *(const v16bf*)(pB3);

        pA0 += 32; pA1 += 32;
        pB0 += 32; pB1 += 32; pB2 += 32; pB3 += 32;

        acc[0][0] = __builtin_amdgcn_wmma_f32_16x16x32_bf16(false, aF[0], false, b0, (short)0, acc[0][0], false, false);
        acc[0][1] = __builtin_amdgcn_wmma_f32_16x16x32_bf16(false, aF[0], false, b1, (short)0, acc[0][1], false, false);
        acc[0][2] = __builtin_amdgcn_wmma_f32_16x16x32_bf16(false, aF[0], false, b2, (short)0, acc[0][2], false, false);
        acc[0][3] = __builtin_amdgcn_wmma_f32_16x16x32_bf16(false, aF[0], false, b3, (short)0, acc[0][3], false, false);
        acc[1][0] = __builtin_amdgcn_wmma_f32_16x16x32_bf16(false, aF[1], false, b0, (short)0, acc[1][0], false, false);
        acc[1][1] = __builtin_amdgcn_wmma_f32_16x16x32_bf16(false, aF[1], false, b1, (short)0, acc[1][1], false, false);
        acc[1][2] = __builtin_amdgcn_wmma_f32_16x16x32_bf16(false, aF[1], false, b2, (short)0, acc[1][2], false, false);
        acc[1][3] = __builtin_amdgcn_wmma_f32_16x16x32_bf16(false, aF[1], false, b3, (short)0, acc[1][3], false, false);
    }

    // Epilogue: D layout -> VGPR r is row (r + 8*half), lane lm is column.
    #pragma unroll
    for (int j = 0; j < 4; ++j) {
        const int n = n0 + j * 16 + lm;
        const float bv = bias ? bias[n] : 0.f;
        #pragma unroll
        for (int s = 0; s < 2; ++s) {
            #pragma unroll
            for (int r = 0; r < 8; ++r) {
                const int m = m0 + s * 16 + half * 8 + r;
                const float v = acc[s][j][r] + bv;
                if (outF) outF[(long)m * ldo + n] = v;
                if (outB) outB[(long)m * ldob + obTot + n] = (__bf16)v;
            }
        }
    }
}

// ---------------------------------------------------------------------------
// Encoder post-GEMM: per-token LayerNorm(2048), split mu/logvar, KL partials,
// DAG softplus mix:  z_causal = zc + dag @ zc  -> bf16 [C][M][CD]
// One block (256 threads) per token.
// ---------------------------------------------------------------------------
__global__ void enc_ln_causal(const float* __restrict__ Y,       // [M, 2048]
                              const float* __restrict__ g,       // [2048]
                              const float* __restrict__ bt,      // [2048]
                              const float* __restrict__ dag_w,   // [16,16]
                              __bf16* __restrict__ zcb,          // [C][M][64]
                              float* __restrict__ klpart) {      // [M]
    __shared__ float red[256];
    __shared__ float shmu[Lc];
    __shared__ float shdag[Cc * Cc];

    const int row = blockIdx.x;
    const int t   = threadIdx.x;
    const float* y = Y + (long)row * TWO_L;

    float v[8];
    float s = 0.f, sq = 0.f;
    #pragma unroll
    for (int k = 0; k < 8; ++k) {
        v[k] = y[t + 256 * k];
        s  += v[k];
        sq += v[k] * v[k];
    }
    const float sum  = blockReduce256(s,  red);
    const float sums = blockReduce256(sq, red);
    const float mean = sum * (1.f / TWO_L);
    const float var  = sums * (1.f / TWO_L) - mean * mean;
    const float rs   = rsqrtf(var + LN_EPS);

    float mu[4], klacc = 0.f;
    #pragma unroll
    for (int k = 0; k < 8; ++k) {
        const int i = t + 256 * k;
        const float h = (v[k] - mean) * rs * g[i] + bt[i];
        if (k < 4) { mu[k] = h; shmu[i] = h; }
        else {
            const float lv = h;
            const float m_ = mu[k - 4];
            klacc += 1.f + lv - m_ * m_ - expf(lv);
        }
    }
    // DAG: strictly-lower softplus
    {
        const int i = t >> 4, j = t & 15;
        shdag[t] = (i > j) ? softplusf(dag_w[t]) : 0.f;
    }
    __syncthreads();

    #pragma unroll
    for (int k = 0; k < 4; ++k) {
        const int idx = t + 256 * k;
        const int c = idx >> 6, d = idx & 63;
        float zv = shmu[idx];
        #pragma unroll
        for (int j = 0; j < Cc; ++j)
            zv += shdag[c * Cc + j] * shmu[j * CDc + d];
        zcb[((long)c * Mc + row) * CDc + d] = (__bf16)zv;
    }

    const float kls = blockReduce256(klacc, red);
    if (t == 0) klpart[row] = kls;
}

// ---------------------------------------------------------------------------
// Per-concept LN(64) + ReLU over rows of t1 [C*M, 64]; one wave per row.
// ---------------------------------------------------------------------------
__global__ void concept_ln_relu(const float* __restrict__ t1,   // [C*M, 64]
                                const float* __restrict__ cg,   // [C,64]
                                const float* __restrict__ cbt,  // [C,64]
                                __bf16* __restrict__ t1b) {     // [C*M, 64]
    const int lane = threadIdx.x & 31;
    const long R = ((long)blockIdx.x * blockDim.x + threadIdx.x) >> 5;
    const int c = (int)(R >> 14);                  // R / M (M = 16384)
    const float* rowp = t1 + R * CDc;

    const float e0 = rowp[lane];
    const float e1 = rowp[lane + 32];
    float s  = e0 + e1;
    float sq = e0 * e0 + e1 * e1;
    #pragma unroll
    for (int m = 16; m > 0; m >>= 1) {
        s  += __shfl_xor(s,  m, 32);
        sq += __shfl_xor(sq, m, 32);
    }
    const float mean = s * (1.f / CDc);
    const float var  = sq * (1.f / CDc) - mean * mean;
    const float rs   = rsqrtf(var + LN_EPS);

    const float* gg = cg  + c * CDc;
    const float* bb = cbt + c * CDc;
    const float o0 = fmaxf(0.f, (e0 - mean) * rs * gg[lane]      + bb[lane]);
    const float o1 = fmaxf(0.f, (e1 - mean) * rs * gg[lane + 32] + bb[lane + 32]);
    t1b[R * CDc + lane]      = (__bf16)o0;
    t1b[R * CDc + lane + 32] = (__bf16)o1;
}

// ---------------------------------------------------------------------------
// Decoder LN(1024) + ReLU; one block per token.
// ---------------------------------------------------------------------------
__global__ void dec_ln_relu(const float* __restrict__ P,   // [M,1024]
                            const float* __restrict__ g,
                            const float* __restrict__ bt,
                            __bf16* __restrict__ outb) {   // [M,1024]
    __shared__ float red[256];
    const int row = blockIdx.x;
    const int t = threadIdx.x;
    const float* y = P + (long)row * Lc;

    float v[4]; float s = 0.f, sq = 0.f;
    #pragma unroll
    for (int k = 0; k < 4; ++k) {
        v[k] = y[t + 256 * k];
        s += v[k]; sq += v[k] * v[k];
    }
    const float sum  = blockReduce256(s,  red);
    const float sums = blockReduce256(sq, red);
    const float mean = sum * (1.f / Lc);
    const float var  = sums * (1.f / Lc) - mean * mean;
    const float rs   = rsqrtf(var + LN_EPS);
    #pragma unroll
    for (int k = 0; k < 4; ++k) {
        const int i = t + 256 * k;
        const float h = fmaxf(0.f, (v[k] - mean) * rs * g[i] + bt[i]);
        outb[(long)row * Lc + i] = (__bf16)h;
    }
}

// ---------------------------------------------------------------------------
// Recon-loss partials over M*D elements -> part[gridDim.x]
// ---------------------------------------------------------------------------
__global__ void recon_partial(const float* __restrict__ xr,
                              const float* __restrict__ x,
                              float* __restrict__ part, long n) {
    __shared__ float red[256];
    float acc = 0.f;
    long i = (long)blockIdx.x * blockDim.x + threadIdx.x;
    const long stride = (long)gridDim.x * blockDim.x;
    for (; i < n; i += stride) {
        const float d = xr[i] - x[i];
        acc += d * d;
    }
    const float s = blockReduce256(acc, red);
    if (threadIdx.x == 0) part[blockIdx.x] = s;
}

// ---------------------------------------------------------------------------
// Finalize the 4 scalar losses. Single block of 256 threads.
// trace(expm(dag)) == C exactly (strictly-lower-triangular => nilpotent).
// ---------------------------------------------------------------------------
__global__ void finalize_losses(const float* __restrict__ reconpart,  // [1024]
                                const float* __restrict__ klpart,     // [16384]
                                const float* __restrict__ dag_w,      // [256]
                                float* __restrict__ outs) {           // [4]
    __shared__ float red[256];
    const int t = threadIdx.x;

    float ra = 0.f;
    #pragma unroll
    for (int k = 0; k < 4; ++k) ra += reconpart[t + 256 * k];
    const float reconSum = blockReduce256(ra, red);

    float ka = 0.f;
    for (int k = 0; k < 64; ++k) ka += klpart[t + 256 * k];
    const float klSum = blockReduce256(ka, red);

    const int i = t >> 4, j = t & 15;
    const float dv = (i > j) ? softplusf(dag_w[t]) : 0.f;
    const float dagSum = blockReduce256(dv, red);

    if (t == 0) {
        const float inv = 1.f / ((float)Mc * (float)Dc);   // == 1/(M*L) too
        const float recon = reconSum * inv;
        const float kl    = -0.5f * klSum * inv;
        const float dagl  = dagSum + (float)Cc;            // trace(expm(dag)) == C
        outs[0] = recon + ALPHA * kl + dagl;               // total
        outs[1] = kl;
        outs[2] = recon;
        outs[3] = dagl;
    }
}

// ---------------------------------------------------------------------------
// Host-side launch
// ---------------------------------------------------------------------------
static inline void launch_gemm(const __bf16* A, const __bf16* W, const float* bias,
                               float* outF, __bf16* outB,
                               int M, int N, int K, int ldo, int ldob, int obOff,
                               long sA, long sW, long sB, long sO, int obBatchOff,
                               int batches, hipStream_t stream) {
    const int waves = (M >> 5) * (N >> 6);
    dim3 grid((waves + 7) / 8, batches);
    gemm_bf16_wmma<<<grid, 256, 0, stream>>>(A, W, bias, outF, outB, M, N, K,
                                             ldo, ldob, obOff, sA, sW, sB, sO, obBatchOff);
}

extern "C" void kernel_launch(void* const* d_in, const int* in_sizes, int n_in,
                              void* d_out, int out_size, void* d_ws, size_t ws_size,
                              hipStream_t stream) {
    const float* x      = (const float*)d_in[0];
    const float* enc_w  = (const float*)d_in[1];
    const float* enc_b  = (const float*)d_in[2];
    const float* enc_g  = (const float*)d_in[3];
    const float* enc_bt = (const float*)d_in[4];
    const float* dag_w  = (const float*)d_in[5];
    const float* cw1    = (const float*)d_in[6];
    const float* cb1    = (const float*)d_in[7];
    const float* cg     = (const float*)d_in[8];
    const float* cbt    = (const float*)d_in[9];
    const float* cw2    = (const float*)d_in[10];
    const float* cb2    = (const float*)d_in[11];
    const float* dec_w1 = (const float*)d_in[12];
    const float* dec_b1 = (const float*)d_in[13];
    const float* dec_g  = (const float*)d_in[14];
    const float* dec_bt = (const float*)d_in[15];
    const float* dec_w2 = (const float*)d_in[16];
    const float* dec_b2 = (const float*)d_in[17];

    char* ws = (char*)d_ws;
    // Workspace layout (bytes). Y1 region (128MB) is reused for t1 (f32) and d_pre (f32);
    // zcb region (32MB) is reused for the decoder bf16 activation.
    const size_t OFF_Y1    = 0;                                  // f32 [M,2048]
    const size_t OFF_T1    = 0;                                  // f32 [C*M,64] (reuse)
    const size_t OFF_DPRE  = 67108864;                           // f32 [M,1024] (reuse)
    const size_t OFF_XB    = 134217728;                          // bf16 [M,1024]
    const size_t OFF_ENCWB = OFF_XB    + (size_t)Mc * Dc * 2;    // bf16 [2048,1024]
    const size_t OFF_DW1B  = OFF_ENCWB + (size_t)TWO_L * Dc * 2; // bf16 [1024,1024]
    const size_t OFF_DW2B  = OFF_DW1B  + (size_t)Lc * Lc * 2;    // bf16 [1024,1024]
    const size_t OFF_CW1B  = OFF_DW2B  + (size_t)Dc * Lc * 2;    // bf16 [16,64,64]
    const size_t OFF_CW2B  = OFF_CW1B  + (size_t)Cc * CDc * CDc * 2;
    const size_t OFF_ZCB   = OFF_CW2B  + (size_t)Cc * CDc * CDc * 2; // bf16 [C,M,64]; reused as db
    const size_t OFF_T1B   = OFF_ZCB   + (size_t)Cc * Mc * CDc * 2;  // bf16 [C,M,64]
    const size_t OFF_ZTB   = OFF_T1B   + (size_t)Cc * Mc * CDc * 2;  // bf16 [M,1024]
    const size_t OFF_KL    = OFF_ZTB   + (size_t)Mc * Lc * 2;        // f32 [M]
    const size_t OFF_RP    = OFF_KL    + (size_t)Mc * 4;             // f32 [1024]

    float*  Y1    = (float*) (ws + OFF_Y1);
    float*  t1    = (float*) (ws + OFF_T1);
    float*  dpre  = (float*) (ws + OFF_DPRE);
    __bf16* xb    = (__bf16*)(ws + OFF_XB);
    __bf16* encwb = (__bf16*)(ws + OFF_ENCWB);
    __bf16* dw1b  = (__bf16*)(ws + OFF_DW1B);
    __bf16* dw2b  = (__bf16*)(ws + OFF_DW2B);
    __bf16* cw1b  = (__bf16*)(ws + OFF_CW1B);
    __bf16* cw2b  = (__bf16*)(ws + OFF_CW2B);
    __bf16* zcb   = (__bf16*)(ws + OFF_ZCB);
    __bf16* db    = (__bf16*)(ws + OFF_ZCB);   // reuse (zcb consumed before db written)
    __bf16* t1b   = (__bf16*)(ws + OFF_T1B);
    __bf16* ztb   = (__bf16*)(ws + OFF_ZTB);
    float*  klp   = (float*) (ws + OFF_KL);
    float*  rpp   = (float*) (ws + OFF_RP);

    float* xrec = (float*)d_out;
    float* outs = xrec + (size_t)Mc * Dc;

    // 1) bf16 casts
    auto cvt = [&](const float* in, __bf16* out, long n) {
        int blocks = (int)((n + 255) / 256); if (blocks > 8192) blocks = 8192;
        cvt_f32_bf16<<<blocks, 256, 0, stream>>>(in, out, n);
    };
    cvt(x,      xb,    (long)Mc * Dc);
    cvt(enc_w,  encwb, (long)TWO_L * Dc);
    cvt(dec_w1, dw1b,  (long)Lc * Lc);
    cvt(dec_w2, dw2b,  (long)Dc * Lc);
    cvt(cw1,    cw1b,  (long)Cc * CDc * CDc);
    cvt(cw2,    cw2b,  (long)Cc * CDc * CDc);

    // 2) encoder GEMM: Y1 = x @ enc_w^T + enc_b   [M,2048]
    launch_gemm(xb, encwb, enc_b, Y1, nullptr, Mc, TWO_L, Dc, TWO_L, 0, 0,
                0, 0, 0, 0, 0, 1, stream);

    // 3) LN(2048) + KL partials + DAG causal mix -> zcb bf16 [C][M][64]
    enc_ln_causal<<<Mc, 256, 0, stream>>>(Y1, enc_g, enc_bt, dag_w, zcb, klp);

    // 4) concept GEMM #1 (batched over C): t1[c] = zc[c] @ cw1[c]^T + cb1[c]
    launch_gemm(zcb, cw1b, cb1, t1, nullptr, Mc, CDc, CDc, CDc, 0, 0,
                (long)Mc * CDc, (long)CDc * CDc, CDc, (long)Mc * CDc, 0, Cc, stream);

    // 5) LN(64) + ReLU per (c, token) row
    {
        const long rows = (long)Cc * Mc;           // 262144, 8 rows per block
        concept_ln_relu<<<(int)(rows / 8), 256, 0, stream>>>(t1, cg, cbt, t1b);
    }

    // 6) concept GEMM #2: z_t[:, c*64 + d] = t1b[c] @ cw2[c]^T + cb2[c]  (bf16 scatter)
    launch_gemm(t1b, cw2b, cb2, nullptr, ztb, Mc, CDc, CDc, 0, Lc, 0,
                (long)Mc * CDc, (long)CDc * CDc, CDc, 0, CDc, Cc, stream);

    // 7) decoder GEMM #1: dpre = z_t @ dec_w1^T + dec_b1   [M,1024]
    launch_gemm(ztb, dw1b, dec_b1, dpre, nullptr, Mc, Lc, Lc, Lc, 0, 0,
                0, 0, 0, 0, 0, 1, stream);

    // 8) LN(1024) + ReLU -> db bf16
    dec_ln_relu<<<Mc, 256, 0, stream>>>(dpre, dec_g, dec_bt, db);

    // 9) decoder GEMM #2: x_recon = db @ dec_w2^T + dec_b2 -> d_out
    launch_gemm(db, dw2b, dec_b2, xrec, nullptr, Mc, Dc, Lc, Dc, 0, 0,
                0, 0, 0, 0, 0, 1, stream);

    // 10) recon partials (1024 blocks)
    recon_partial<<<1024, 256, 0, stream>>>(xrec, x, rpp, (long)Mc * Dc);

    // 11) finalize scalars: [total, kl, recon, dag]
    finalize_losses<<<1, 256, 0, stream>>>(rpp, klp, dag_w, outs);

    (void)in_sizes; (void)n_in; (void)out_size; (void)ws_size;
}
